// GPTv2_429496730287
// MI455X (gfx1250) — matrix operand order repaired and checked by reading, not compiled, not run on hardware
//
#include <hip/hip_runtime.h>

// ---------------------------------------------------------------------------
// GPT forward pass for MI455X (gfx1250).
// All matmuls via v_wmma_f32_16x16x32_bf16; A-tiles staged with the Tensor
// Data Mover (tensor_load_to_lds + s_wait_tensorcnt) when available; all
// WMMA fragments loaded as contiguous 128-bit chunks (layout-aware tiling).
// ---------------------------------------------------------------------------

#define DEV __device__ __forceinline__

static constexpr int Vv = 32000;
static constexpr int Cc = 1024;
static constexpr int Hh = 16;
static constexpr int Dd = 64;
static constexpr int Ll = 6;
static constexpr int Bb = 4;
static constexpr int Tt = 1024;
static constexpr int Mm = Bb * Tt;     // 4096 activation rows
static constexpr int FF = 4 * Cc;      // 4096
static constexpr float QK_SCALE = 0.03125f;  // C^-0.5 (reference scales by full n_embed)

typedef __attribute__((ext_vector_type(16))) __bf16 v16bf;
typedef __attribute__((ext_vector_type(8)))  float  v8f;
typedef unsigned int u32x4 __attribute__((ext_vector_type(4)));
typedef int          i32x4 __attribute__((ext_vector_type(4)));
typedef int          i32x8 __attribute__((ext_vector_type(8)));

#if __has_builtin(__builtin_amdgcn_tensor_load_to_lds)
#define HAVE_TDM 1
#else
#define HAVE_TDM 0
#endif

DEV unsigned short f2bf(float f) {           // cheap round-to-nearest-up
  union { float f; unsigned u; } c; c.f = f;
  return (unsigned short)((c.u + 0x8000u) >> 16);
}
DEV float bf2f(unsigned short h) {
  union { unsigned u; float f; } c; c.u = ((unsigned)h) << 16;
  return c.f;
}
DEV unsigned pack2(float a, float b) {
  return (unsigned)f2bf(a) | ((unsigned)f2bf(b) << 16);
}
DEV void wait_tensorcnt0() {
#if __has_builtin(__builtin_amdgcn_s_wait_tensorcnt)
  __builtin_amdgcn_s_wait_tensorcnt(0);
#else
  asm volatile("s_wait_tensorcnt 0x0" ::: "memory");
#endif
}

// reinterpret two 16-byte chunks as one WMMA 16-element bf16 fragment
union FragCast { uint4 u[2]; v16bf v; };
static_assert(sizeof(v16bf) == 32, "v16bf must be 32 bytes");

// ---------------------------------------------------------------------------
// Generic tiled GEMM: out[M,N] = act(A[M,K] @ W[K,N] + bias) (+ residual)
// BM=128, BN=64, BK=32; 256 threads = 8 wave32, one 16x64 strip per wave.
// A tile: row-major LDS, 80B row stride (16 dwords data + 4 dwords pad) ->
//   staged by TDM (bf16 A) or vectorized copies; fragments = 2x ds_load_b128.
// W tile: transposed in LDS (Bs_t[n][k]); fragments = 2x ds_load_b128.
// OUT_MODE 0: linear.  1: [B,H,T,D] head layout.  2: [B,H,D,T] (transposed V).
// ---------------------------------------------------------------------------
template<bool A_BF16, bool BIAS, bool RELU, bool RES, bool OUT_BF16, int OUT_MODE>
__global__ __launch_bounds__(256)
void gemm_bf16_wmma(const void* __restrict__ Ap, const float* __restrict__ W,
                    const float* __restrict__ bias, const float* __restrict__ res,
                    void* __restrict__ outv, int M, int N, int K,
                    long long zA, long long zW, long long zOut)
{
  __shared__ __align__(16) unsigned short As[128][40];    // 10 KB
  __shared__ __align__(16) unsigned short Bs_t[64][40];   //  5 KB

  const int tid  = threadIdx.x;
  const int wv   = tid >> 5;
  const int lane = tid & 31;
  const int half = lane >> 4;
  const int l16  = lane & 15;
  const int m0   = blockIdx.y * 128;
  const int n0   = blockIdx.x * 64;
  const long long z = blockIdx.z;

  const float*          Af = (const float*)Ap;
  const unsigned short* Ab = (const unsigned short*)Ap;
  if (A_BF16) Ab += z * zA; else Af += z * zA;
  W += z * zW;

  const int nB = tid & 63;     // B-stage: column
  const int kg = tid >> 6;     // B-stage: k group

  v8f acc[4];
#pragma unroll
  for (int i = 0; i < 4; ++i)
#pragma unroll
    for (int r = 0; r < 8; ++r) acc[i][r] = 0.f;

  const int nk = K >> 5;
  for (int kt = 0; kt < nk; ++kt) {
    const int k0 = kt << 5;

    // ---- stage A tile (128x32 bf16, row-major, 80B row stride) ----
    if (A_BF16) {
#if HAVE_TDM
      if (wv == 0) {
        const unsigned long long ga =
            (unsigned long long)(const void*)(Ab + (size_t)m0 * K + k0);
        const unsigned ldsOff =
            (unsigned)(unsigned long long)(const void*)&As[0][0];
        u32x4 g0 = { 1u,                    // count=1 (valid user D#)
                     ldsOff,                // lds_addr (bytes)
                     (unsigned)ga,          // global_addr[31:0]
                     ((unsigned)((ga >> 32) & 0x01FFFFFFull)) | 0x80000000u };
        i32x8 g1 = { (int)((1u << 16)       // data_size = 2 bytes
                   | (1u << 20)             // pad_enable
                   | (3u << 22)             // pad_interval = 16 dwords
                   | (3u << 25)),           // pad_amount   = 4 dwords
                     (int)((K & 0xFFFF) << 16),            // tensor_dim0 lo
                     (int)((K >> 16) | ((M & 0xFFFF) << 16)),
                     (int)((M >> 16) | (32 << 16)),        // tile_dim0 = 32
                     128,                                  // tile_dim1 = 128
                     K,                                    // dim0_stride lo
                     0, 0 };
        i32x4 gz = { 0, 0, 0, 0 };
#if __clang_major__ >= 23
        i32x8 gz8 = { 0, 0, 0, 0, 0, 0, 0, 0 };
        __builtin_amdgcn_tensor_load_to_lds(g0, g1, gz, gz, gz8, 0);
#else
        __builtin_amdgcn_tensor_load_to_lds(g0, g1, gz, gz, 0);
#endif
      }
      wait_tensorcnt0();
#else
#pragma unroll
      for (int i = tid; i < 512; i += 256) {      // 16-byte blocks
        const int rr = i >> 2, c8 = (i & 3) * 8;
        *(uint4*)&As[rr][c8] =
            *(const uint4*)&Ab[(size_t)(m0 + rr) * K + k0 + c8];
      }
#endif
    } else {
#pragma unroll
      for (int i = tid; i < 1024; i += 256) {     // 4-float blocks -> 4 bf16
        const int rr = i >> 3, c4 = (i & 7) * 4;
        const float4 f = *(const float4*)&Af[(size_t)(m0 + rr) * K + k0 + c4];
        uint2 p2; p2.x = pack2(f.x, f.y); p2.y = pack2(f.z, f.w);
        *(uint2*)&As[rr][c4] = p2;
      }
    }

    // ---- stage W tile transposed: Bs_t[n][k], packed ds_store_b64 ----
#pragma unroll
    for (int ph = 0; ph < 2; ++ph) {
      const int kk0 = ph * 16 + kg * 4;
      const float w0 = W[(size_t)(k0 + kk0 + 0) * N + n0 + nB];
      const float w1 = W[(size_t)(k0 + kk0 + 1) * N + n0 + nB];
      const float w2 = W[(size_t)(k0 + kk0 + 2) * N + n0 + nB];
      const float w3 = W[(size_t)(k0 + kk0 + 3) * N + n0 + nB];
      uint2 p2; p2.x = pack2(w0, w1); p2.y = pack2(w2, w3);
      *(uint2*)&Bs_t[nB][kk0] = p2;
    }
    if (kt + 1 < nk)   // global_prefetch_b8 hint for next weight tile
      __builtin_prefetch(&W[(size_t)(k0 + 32) * N + n0 + nB], 0, 1);
    __syncthreads();

    // ---- fragments: contiguous 2x b128 each ----
    FragCast fa;
    const uint4* arow = (const uint4*)&As[wv * 16 + l16][0];
    fa.u[0] = arow[half];        // K = half*8 .. +7
    fa.u[1] = arow[half + 2];    // K = 16+half*8 .. +7
#pragma unroll
    for (int nt = 0; nt < 4; ++nt) {
      FragCast fb;
      const uint4* brow = (const uint4*)&Bs_t[nt * 16 + l16][0];
      fb.u[0] = brow[half];          // K = half*8 .. +7
      fb.u[1] = brow[half + 2];      // K = 16+half*8 .. +7
      acc[nt] = __builtin_amdgcn_wmma_f32_16x16x32_bf16(
          false, fa.v, false, fb.v, (short)0, acc[nt], false, false);
    }
    __syncthreads();
  }

  // ---- epilogue: D layout row = r + half*8, col = lane&15 ----
#pragma unroll
  for (int nt = 0; nt < 4; ++nt) {
#pragma unroll
    for (int r = 0; r < 8; ++r) {
      const int mrow = m0 + wv * 16 + half * 8 + r;
      const int ncol = n0 + nt * 16 + l16;
      float v = acc[nt][r];
      if (BIAS) v += bias[ncol];
      if (RES)  v += res[(size_t)mrow * N + ncol];
      if (RELU) v = fmaxf(v, 0.f);
      size_t off;
      if (OUT_MODE == 0)      off = (size_t)mrow * N + ncol;
      else if (OUT_MODE == 1) off = (size_t)(mrow >> 10) * ((size_t)Hh * Tt * Dd)
                                  + (size_t)(mrow & (Tt - 1)) * Dd + ncol;
      else                    off = (size_t)(mrow >> 10) * ((size_t)Hh * Tt * Dd)
                                  + (size_t)ncol * Tt + (mrow & (Tt - 1));
      if (OUT_BF16) ((unsigned short*)outv)[(size_t)(z * zOut) + off] = f2bf(v);
      else          ((float*)outv)[(size_t)(z * zOut) + off] = v;
    }
  }
}

// ---------------------------------------------------------------------------
// Fused causal attention, one wave32 per (b, h, 16-query-row tile).
// S = (q @ k^T)*scale + pk @ pq^T; softmax strip kept in LDS (unnormalized
// bf16 exps, 1/rowsum folded into output); O = P @ V^T-layout. All fragments
// are contiguous 128-bit loads; all matmuls WMMA.
// v is stored transposed [B,H,D,T] so its B-fragments are contiguous.
// ---------------------------------------------------------------------------
__global__ __launch_bounds__(32)
void attn_kernel(const unsigned short* __restrict__ q, const unsigned short* __restrict__ k,
                 const unsigned short* __restrict__ vT, const unsigned short* __restrict__ pk,
                 const unsigned short* __restrict__ pq, unsigned short* __restrict__ o)
{
  __shared__ __align__(16) unsigned short prob[16][Tt + 8];  // 33 KB strip
  __shared__ float invsum[16];

  const int lane = threadIdx.x;
  const int half = lane >> 4;
  const int l16  = lane & 15;
  const int qt = blockIdx.x, h = blockIdx.y, b = blockIdx.z;
  const int t0 = qt * 16;

  const size_t bh = ((size_t)b * Hh + h) * Tt * Dd;
  const unsigned short* qh  = q  + bh;
  const unsigned short* kh  = k  + bh;
  const unsigned short* vh  = vT + bh;                 // [D][T] within head
  const unsigned short* pkh = pk + (size_t)h * Tt * Dd;
  const unsigned short* pqh = pq + (size_t)h * Tt * Dd;

  // A-fragments for the 16 query rows: contiguous b128 pairs
  const uint4* qrow  = (const uint4*)(qh  + (size_t)(t0 + l16) * Dd);
  const uint4* pkrow = (const uint4*)(pkh + (size_t)(t0 + l16) * Dd);
  FragCast aq0, aq1, ap0, ap1;
  aq0.u[0] = qrow[half];      aq0.u[1] = qrow[half + 2];       // d 0..31
  aq1.u[0] = qrow[4 + half];  aq1.u[1] = qrow[6 + half];       // d 32..63
  ap0.u[0] = pkrow[half];     ap0.u[1] = pkrow[half + 2];
  ap1.u[0] = pkrow[4 + half]; ap1.u[1] = pkrow[6 + half];

  const int ntile = qt + 1;                 // causal: key tiles 0..qt
  for (int st = 0; st < ntile; ++st) {
    const int s0 = st * 16;
    const uint4* krow  = (const uint4*)(kh  + (size_t)(s0 + l16) * Dd);
    const uint4* pqrow = (const uint4*)(pqh + (size_t)(s0 + l16) * Dd);
    FragCast bk0, bk1, bq0, bq1;
    bk0.u[0] = krow[half];      bk0.u[1] = krow[half + 2];    // d 0..31
    bk1.u[0] = krow[4 + half];  bk1.u[1] = krow[6 + half];    // d 32..63
    bq0.u[0] = pqrow[half];     bq0.u[1] = pqrow[half + 2];
    bq1.u[0] = pqrow[4 + half]; bq1.u[1] = pqrow[6 + half];

    v8f s1, s2;
#pragma unroll
    for (int r = 0; r < 8; ++r) { s1[r] = 0.f; s2[r] = 0.f; }
    s1 = __builtin_amdgcn_wmma_f32_16x16x32_bf16(false, aq0.v, false, bk0.v, (short)0, s1, false, false);
    s1 = __builtin_amdgcn_wmma_f32_16x16x32_bf16(false, aq1.v, false, bk1.v, (short)0, s1, false, false);
    s2 = __builtin_amdgcn_wmma_f32_16x16x32_bf16(false, ap0.v, false, bq0.v, (short)0, s2, false, false);
    s2 = __builtin_amdgcn_wmma_f32_16x16x32_bf16(false, ap1.v, false, bq1.v, (short)0, s2, false, false);
#pragma unroll
    for (int r = 0; r < 8; ++r) {
      const int mrow = r + half * 8;
      const int tg = t0 + mrow, sg = s0 + l16;
      float val = s1[r] * QK_SCALE + s2[r];
      if (sg > tg) val = -1.0e30f;          // causal mask
      prob[mrow][sg] = f2bf(val);
    }
  }

  const int s_len = t0 + 16;
  const int s_pad = (ntile & 1) ? (s_len + 16) : s_len;   // pad to 32 for P@V
  if (ntile & 1) {
    for (int i = lane; i < 256; i += 32) prob[i >> 4][s_len + (i & 15)] = 0;
  }
  __syncthreads();

  // softmax stats: two lanes per row; keep unnormalized exps in LDS
  {
    const int row = lane >> 1, j = lane & 1;
    float mx = -3.0e38f;
    for (int s = j; s < s_len; s += 2) mx = fmaxf(mx, bf2f(prob[row][s]));
    mx = fmaxf(mx, __shfl_xor(mx, 1, 32));
    float sum = 0.f;
    for (int s = j; s < s_len; s += 2) {
      const float ev = __expf(bf2f(prob[row][s]) - mx);
      sum += ev;
      prob[row][s] = f2bf(ev);
    }
    sum += __shfl_xor(sum, 1, 32);
    if (j == 0) invsum[row] = 1.0f / sum;
  }
  __syncthreads();

  // O = P @ V  (P A-fragments and V^T B-fragments both contiguous)
  v8f oacc[4];
#pragma unroll
  for (int i = 0; i < 4; ++i)
#pragma unroll
    for (int r = 0; r < 8; ++r) oacc[i][r] = 0.f;

  const uint4* prow = (const uint4*)&prob[l16][0];
  for (int sb = 0; sb < s_pad; sb += 32) {
    FragCast fp;
    fp.u[0] = prow[(sb >> 3) + half];       // s = sb + half*8 ..
    fp.u[1] = prow[(sb >> 3) + half + 2];   // s = sb + 16 + half*8 ..
#pragma unroll
    for (int dt = 0; dt < 4; ++dt) {
      const uint4* vrow = (const uint4*)(vh + (size_t)(dt * 16 + l16) * Tt);
      FragCast fv;
      const int vi = (sb >> 3) + half;
      fv.u[0] = vrow[vi];
      fv.u[1] = vrow[vi + 2];
      oacc[dt] = __builtin_amdgcn_wmma_f32_16x16x32_bf16(
          false, fp.v, false, fv.v, (short)0, oacc[dt], false, false);
    }
  }
  // heads concatenated: o[b, t, h*64 + d], scaled by 1/rowsum
#pragma unroll
  for (int dt = 0; dt < 4; ++dt)
#pragma unroll
    for (int r = 0; r < 8; ++r) {
      const int mrow = r + half * 8;
      const int tg = t0 + mrow;
      const float val = oacc[dt][r] * invsum[mrow];
      o[((size_t)b * Tt + tg) * Cc + h * Dd + dt * 16 + l16] = f2bf(val);
    }
}

// ---------------------------------------------------------------------------
// LayerNorm row kernel: fp32 in -> bf16 out (feeds the GEMMs)
// ---------------------------------------------------------------------------
__global__ __launch_bounds__(256)
void ln_kernel(const float* __restrict__ x, const float* __restrict__ w,
               const float* __restrict__ bb, unsigned short* __restrict__ y, int Cn)
{
  __shared__ float red[256];
  const int m = blockIdx.x, tid = threadIdx.x;
  const float* xr = x + (size_t)m * Cn;

  float s = 0.f;
  for (int c = tid; c < Cn; c += 256) s += xr[c];
  red[tid] = s; __syncthreads();
  for (int st = 128; st > 0; st >>= 1) { if (tid < st) red[tid] += red[tid + st]; __syncthreads(); }
  const float mean = red[0] / Cn;
  __syncthreads();

  float vs = 0.f;
  for (int c = tid; c < Cn; c += 256) { const float d = xr[c] - mean; vs += d * d; }
  red[tid] = vs; __syncthreads();
  for (int st = 128; st > 0; st >>= 1) { if (tid < st) red[tid] += red[tid + st]; __syncthreads(); }
  const float rstd = rsqrtf(red[0] / Cn + 1e-5f);

  unsigned short* yr = y + (size_t)m * Cn;
  for (int c = tid * 2; c < Cn; c += 512) {
    unsigned p = pack2((xr[c] - mean) * rstd * w[c] + bb[c],
                       (xr[c + 1] - mean) * rstd * w[c + 1] + bb[c + 1]);
    *(unsigned*)&yr[c] = p;
  }
}

// Embedding gather: x[m, :] = tok_tab[idx[m], :]
__global__ __launch_bounds__(256)
void embed_kernel(const int* __restrict__ idx, const float* __restrict__ tab,
                  float* __restrict__ x)
{
  const int m = blockIdx.x;
  const int tok = idx[m];
  for (int c = threadIdx.x * 4; c < Cc; c += 1024)
    *(float4*)&x[(size_t)m * Cc + c] = *(const float4*)&tab[(size_t)tok * Cc + c];
}

// ---------------------------------------------------------------------------
extern "C" void kernel_launch(void* const* d_in, const int* in_sizes, int n_in,
                              void* d_out, int out_size, void* d_ws, size_t ws_size,
                              hipStream_t stream)
{
  (void)in_sizes; (void)n_in; (void)out_size; (void)ws_size;

  const int*   idx     = (const int*)  d_in[0];
  const float* tok_tab = (const float*)d_in[1];
  const float* Wk      = (const float*)d_in[2];
  const float* Wq      = (const float*)d_in[3];
  const float* Wv      = (const float*)d_in[4];
  const float* pos_tab = (const float*)d_in[5];
  const float* proj_w  = (const float*)d_in[6];
  const float* proj_b  = (const float*)d_in[7];
  const float* ln1_w   = (const float*)d_in[8];
  const float* ln1_b   = (const float*)d_in[9];
  const float* ln2_w   = (const float*)d_in[10];
  const float* ln2_b   = (const float*)d_in[11];
  const float* ff_w1   = (const float*)d_in[12];
  const float* ff_b1   = (const float*)d_in[13];
  const float* ff_w2   = (const float*)d_in[14];
  const float* ff_b2   = (const float*)d_in[15];
  const float* lnf_w   = (const float*)d_in[16];
  const float* lnf_b   = (const float*)d_in[17];
  const float* lm_w    = (const float*)d_in[18];
  const float* lm_b    = (const float*)d_in[19];

  // workspace carve-out (~92 MB)
  char* p = (char*)d_ws;
  float*          x   = (float*)p;          p += (size_t)Mm * Cc * sizeof(float);
  unsigned short* hb  = (unsigned short*)p; p += (size_t)Mm * Cc * 2;
  unsigned short* qb  = (unsigned short*)p; p += (size_t)Bb * Hh * Tt * Dd * 2;
  unsigned short* kb  = (unsigned short*)p; p += (size_t)Bb * Hh * Tt * Dd * 2;
  unsigned short* vb  = (unsigned short*)p; p += (size_t)Bb * Hh * Tt * Dd * 2;
  unsigned short* pkb = (unsigned short*)p; p += (size_t)Hh * Tt * Dd * 2;
  unsigned short* pqb = (unsigned short*)p; p += (size_t)Hh * Tt * Dd * 2;
  unsigned short* ob  = (unsigned short*)p; p += (size_t)Mm * Cc * 2;
  unsigned short* ffb = (unsigned short*)p; p += (size_t)Mm * FF * 2;

  const long long zqkvW = (long long)Cc * Dd;   // per-head weight stride
  const long long zqkvO = (long long)Tt * Dd;   // per-head output stride
  const long long zposA = (long long)Tt * Cc;   // per-head pos-emb stride

  embed_kernel<<<Mm, 256, 0, stream>>>(idx, tok_tab, x);

  for (int l = 0; l < Ll; ++l) {
    const size_t wl = (size_t)l * Hh * Cc * Dd;
    // LN1 -> bf16
    ln_kernel<<<Mm, 256, 0, stream>>>(x, ln1_w + (size_t)l * Cc, ln1_b + (size_t)l * Cc, hb, Cc);

    // q, k: [B,H,T,D]; v: transposed [B,H,D,T]   (grid.z = head)
    gemm_bf16_wmma<true, false, false, false, true, 1>
      <<<dim3(1, Mm / 128, Hh), 256, 0, stream>>>(hb, Wq + wl, nullptr, nullptr, qb,
                                                  Mm, Dd, Cc, 0, zqkvW, zqkvO);
    gemm_bf16_wmma<true, false, false, false, true, 1>
      <<<dim3(1, Mm / 128, Hh), 256, 0, stream>>>(hb, Wk + wl, nullptr, nullptr, kb,
                                                  Mm, Dd, Cc, 0, zqkvW, zqkvO);
    gemm_bf16_wmma<true, false, false, false, true, 2>
      <<<dim3(1, Mm / 128, Hh), 256, 0, stream>>>(hb, Wv + wl, nullptr, nullptr, vb,
                                                  Mm, Dd, Cc, 0, zqkvW, zqkvO);

    // pk = pe @ Wk, pq = pe @ Wq   (A = fp32 pos_tab, per-head z strides)
    const float* pel = pos_tab + (size_t)l * Hh * Tt * Cc;
    gemm_bf16_wmma<false, false, false, false, true, 0>
      <<<dim3(1, Tt / 128, Hh), 256, 0, stream>>>(pel, Wk + wl, nullptr, nullptr, pkb,
                                                  Tt, Dd, Cc, zposA, zqkvW, zqkvO);
    gemm_bf16_wmma<false, false, false, false, true, 0>
      <<<dim3(1, Tt / 128, Hh), 256, 0, stream>>>(pel, Wq + wl, nullptr, nullptr, pqb,
                                                  Tt, Dd, Cc, zposA, zqkvW, zqkvO);

    // fused causal attention with positional bias
    attn_kernel<<<dim3(Tt / 16, Hh, Bb), 32, 0, stream>>>(qb, kb, vb, pkb, pqb, ob);

    // x = x + o @ proj_w + proj_b
    gemm_bf16_wmma<true, true, false, true, false, 0>
      <<<dim3(Cc / 64, Mm / 128), 256, 0, stream>>>(ob, proj_w + (size_t)l * Cc * Cc,
                                                    proj_b + (size_t)l * Cc, x, x,
                                                    Mm, Cc, Cc, 0, 0, 0);
    // LN2 -> bf16
    ln_kernel<<<Mm, 256, 0, stream>>>(x, ln2_w + (size_t)l * Cc, ln2_b + (size_t)l * Cc, hb, Cc);

    // ff1: relu(h2 @ ff_w1 + ff_b1) -> bf16
    gemm_bf16_wmma<true, true, true, false, true, 0>
      <<<dim3(FF / 64, Mm / 128), 256, 0, stream>>>(hb, ff_w1 + (size_t)l * Cc * FF,
                                                    ff_b1 + (size_t)l * FF, nullptr, ffb,
                                                    Mm, FF, Cc, 0, 0, 0);
    // ff2: x = x + a1 @ ff_w2 + ff_b2
    gemm_bf16_wmma<true, true, false, true, false, 0>
      <<<dim3(Cc / 64, Mm / 128), 256, 0, stream>>>(ffb, ff_w2 + (size_t)l * FF * Cc,
                                                    ff_b2 + (size_t)l * Cc, x, x,
                                                    Mm, Cc, FF, 0, 0, 0);
  }

  // final LN + LM head -> d_out (fp32 logits [B,T,V])
  ln_kernel<<<Mm, 256, 0, stream>>>(x, lnf_w, lnf_b, hb, Cc);
  gemm_bf16_wmma<true, true, false, false, false, 0>
    <<<dim3(Vv / 64, Mm / 128), 256, 0, stream>>>(hb, lm_w, lm_b, nullptr, (float*)d_out,
                                                  Mm, Vv, Cc, 0, 0, 0);
}
